// NystromAttention_20117626814477
// MI455X (gfx1250) — compile-verified, matrix operand order checked by reading, not run on hardware
//
#include <hip/hip_runtime.h>
#include <hip/hip_bf16.h>

// ---------------------------------------------------------------------------
// Nystrom attention for MI455X (gfx1250, wave32, WMMA + TDM).
// B=4, L=4096, D_MODEL=1024, H=16, D=64, m=64 landmarks, 6 NS iterations.
// ---------------------------------------------------------------------------

typedef __attribute__((ext_vector_type(16))) __bf16 v16bf;
typedef __attribute__((ext_vector_type(8)))  __bf16 v8bf;
typedef __attribute__((ext_vector_type(4)))  __bf16 v4bf;
typedef __attribute__((ext_vector_type(8)))  float  v8f;
typedef __attribute__((ext_vector_type(2)))  float  v2f;
typedef __attribute__((ext_vector_type(4)))  unsigned int u32x4;
typedef __attribute__((ext_vector_type(8)))  int    i32x8;
typedef __attribute__((ext_vector_type(4)))  int    i32x4;

#if __has_builtin(__builtin_amdgcn_tensor_load_to_lds)
#define NYS_HAVE_TDM 1
#else
#define NYS_HAVE_TDM 0
#endif

union BFrag { v16bf v; v8bf h[2]; };

// 16x32 bf16 WMMA operand fragment from row-major (NxK, K contig) memory.
// lane<16: row=lane, K in [0..7]+[16..23]; lane>=16: row=lane-16,
// K in [8..15]+[24..31]  (CDNA5 ISA 7.12.2, 16-bit A-matrix layout).
__device__ inline v16bf frag_ld(const __bf16* base, int ld, int r, int half) {
  BFrag f;
  f.h[0] = *(const v8bf*)(base + r * ld + half * 8);
  f.h[1] = *(const v8bf*)(base + r * ld + half * 8 + 16);
  return f.v;
}

__device__ inline v8f wmma_bf16(v16bf a, v16bf b, v8f c) {
  return __builtin_amdgcn_wmma_f32_16x16x32_bf16(false, a, false, b,
                                                 (short)0, c, false, false);
}
__device__ inline v8f wmma_f32(v2f a, v2f b, v8f c) {
  return __builtin_amdgcn_wmma_f32_16x16x4_f32(false, a, false, b,
                                               (short)0, c, false, false);
}
#define ZV8 ((v8f)0.0f)

// ---------------------------------------------------------------------------
// Tensor Data Mover: 2D bf16 tile (rows x COLS) global->LDS with LDS row
// padding generated by the D# pad fields (CDNA5 ISA 08, D# groups 0/1).
// ---------------------------------------------------------------------------
#if NYS_HAVE_TDM
__device__ inline void tdm_load_2d_bf16(unsigned lds_off,
                                        unsigned long long gaddr,
                                        unsigned tile_d0, unsigned tile_d1,
                                        unsigned long long stride0,
                                        unsigned pad_interval_code,
                                        unsigned pad_amount_code) {
  u32x4 g0;
  g0[0] = 1u;                                        // count=1 (user D#)
  g0[1] = lds_off;                                   // LDS byte address
  g0[2] = (unsigned)(gaddr & 0xffffffffu);           // global_addr[31:0]
  g0[3] = (unsigned)((gaddr >> 32) & 0x01ffffffu)    // global_addr[56:32]
          | 0x80000000u;                             // type=2 ("image")
  i32x8 g1;
  unsigned w0 = (1u << 16)                           // data_size = 2 bytes
                | (1u << 20)                         // pad_enable
                | (pad_interval_code << 22) | (pad_amount_code << 25);
  g1[0] = (int)w0;
  g1[1] = (int)((tile_d0 & 0xffffu) << 16);          // tensor_dim0[15:0]
  g1[2] = (int)(((tile_d0 >> 16) & 0xffffu) |        // tensor_dim0[31:16]
                ((tile_d1 & 0xffffu) << 16));        // tensor_dim1[15:0]
  g1[3] = (int)(((tile_d1 >> 16) & 0xffffu) |        // tensor_dim1[31:16]
                ((tile_d0 & 0xffffu) << 16));        // tile_dim0
  g1[4] = (int)(tile_d1 & 0xffffu);                  // tile_dim1 (tile_dim2=0)
  g1[5] = (int)(stride0 & 0xffffffffu);              // dim0_stride[31:0]
  g1[6] = (int)((stride0 >> 32) & 0xffffu);          // dim0_stride[47:32]
  g1[7] = 0;                                         // dim1_stride unused (2D)
  i32x4 gz4 = (i32x4)0;
  i32x8 gz8 = (i32x8)0;
  // 6-arg form (clang tensor builtin): g0, g1, g2, g3, g4, cpol
  __builtin_amdgcn_tensor_load_to_lds(g0, g1, gz4, gz4, gz8, 0);
}
#endif

// Stage `rows` rows of COLS bf16 from global (row stride src_ld elements) into
// LDS at row stride (COLS + COLS/4) elements.  TDM path: wave 0 issues one
// descriptor; fallback: cooperative vector copy by all 256 threads.
template <int COLS>
__device__ inline void stage_rows(__bf16* lds, const __bf16* g, int src_ld,
                                  int rows, int tid) {
#if NYS_HAVE_TDM
  if ((tid >> 5) == 0) {
    const unsigned pi = (COLS == 32) ? 3u : 4u;   // pad every 64B / 128B
    const unsigned pa = (COLS == 32) ? 3u : 7u;   // pad 16B / 32B
    tdm_load_2d_bf16((unsigned)(unsigned long long)(uintptr_t)lds,
                     (unsigned long long)(uintptr_t)g, COLS, (unsigned)rows,
                     (unsigned long long)src_ld, pi, pa);
  }
#else
  const int DLD = COLS + COLS / 4;
  for (int i = tid; i < rows * (COLS / 8); i += 256) {
    int row = i / (COLS / 8), c8 = (i % (COLS / 8)) * 8;
    *(v8bf*)(lds + row * DLD + c8) = *(const v8bf*)(g + (size_t)row * src_ld + c8);
  }
#endif
}

__device__ inline void stage_wait(int tid) {
#if NYS_HAVE_TDM
  if ((tid >> 5) == 0) __builtin_amdgcn_s_wait_tensorcnt(0);
#endif
}

// ---------------------------------------------------------------------------
// f32 -> bf16 bulk conversion (one pass; keeps GEMM inner loops pure bf16)
// ---------------------------------------------------------------------------
__global__ __launch_bounds__(256) void to_bf16(const float* __restrict__ src,
                                               __bf16* __restrict__ dst,
                                               int n4) {
  int i = blockIdx.x * 256 + threadIdx.x;
  if (i < n4) {
    float4 x = ((const float4*)src)[i];
    v4bf o;
    o[0] = (__bf16)x.x; o[1] = (__bf16)x.y;
    o[2] = (__bf16)x.z; o[3] = (__bf16)x.w;
    ((v4bf*)dst)[i] = o;
  }
}

// ---------------------------------------------------------------------------
// GEMM: Y = X @ W^T + bias (X, W pre-converted bf16; W row-major [out,in]).
// MODE 0/1: Q/K proj -> *0.5, mask, bf16 [B,H,L,64]
// MODE 2:   V proj   -> bf16 transposed [B,H,64,L]
// MODE 3:   final    -> f32 [B,L,1024]
// 128x128 tile, K-step 32, 8 waves, TDM double-buffered staging.
// ---------------------------------------------------------------------------
template <int MODE>
__global__ __launch_bounds__(256) void nys_gemm(
    const __bf16* __restrict__ X, const __bf16* __restrict__ Wb,
    const float* __restrict__ bias, const unsigned char* __restrict__ mask,
    __bf16* __restrict__ Yb, float* __restrict__ Yf) {
  __shared__ __align__(16) __bf16 As[2][128][40];
  __shared__ __align__(16) __bf16 Bs[2][128][40];

  const int tid  = threadIdx.x;
  const int lane = tid & 31, wid = tid >> 5;
  const int r    = lane & 15, half = lane >> 4;
  const int wm   = wid & 3,  wn   = wid >> 2;
  const int n0   = blockIdx.x * 128;
  const int o0   = blockIdx.y * 128;

  v8f acc[2][4];
  for (int i = 0; i < 2; ++i)
    for (int j = 0; j < 4; ++j) acc[i][j] = ZV8;

  // prologue: stage k-tile 0 into buffer 0
  stage_rows<32>(&As[0][0][0], X + (size_t)n0 * 1024, 1024, 128, tid);
  stage_rows<32>(&Bs[0][0][0], Wb + (size_t)o0 * 1024, 1024, 128, tid);

  for (int kt = 0; kt < 32; ++kt) {
    const int cur = kt & 1;
    stage_wait(tid);
    __syncthreads();
    if (kt + 1 < 32) {  // overlap TDM of tile kt+1 with compute of tile kt
      int kk = (kt + 1) * 32;
      stage_rows<32>(&As[cur ^ 1][0][0], X + (size_t)n0 * 1024 + kk, 1024, 128, tid);
      stage_rows<32>(&Bs[cur ^ 1][0][0], Wb + (size_t)o0 * 1024 + kk, 1024, 128, tid);
    }
    v16bf fa[2], fb[4];
    for (int i = 0; i < 2; ++i)
      fa[i] = frag_ld(&As[cur][wm * 32 + i * 16][0], 40, r, half);
    for (int j = 0; j < 4; ++j)
      fb[j] = frag_ld(&Bs[cur][wn * 64 + j * 16][0], 40, r, half);
    for (int i = 0; i < 2; ++i)
      for (int j = 0; j < 4; ++j) acc[i][j] = wmma_bf16(fa[i], fb[j], acc[i][j]);
  }

  for (int i = 0; i < 2; ++i)
    for (int j = 0; j < 4; ++j)
      for (int vv = 0; vv < 8; ++vv) {
        int row = n0 + wm * 32 + i * 16 + half * 8 + vv;
        int col = o0 + wn * 64 + j * 16 + r;
        float val = acc[i][j][vv] + bias[col];
        if (MODE <= 1) {  // Q/K: scale by 1/H^0.25 = 0.5 and mask
          int b = row >> 12, l = row & 4095;
          float mf = mask[b * 4096 + l] ? 1.0f : 0.0f;
          val *= 0.5f * mf;
          int h = col >> 6, d = col & 63;
          Yb[(((size_t)b * 16 + h) * 4096 + l) * 64 + d] = (__bf16)val;
        } else if (MODE == 2) {  // V transposed per head: [B,H,D,L]
          int b = row >> 12, l = row & 4095;
          int h = col >> 6, d = col & 63;
          Yb[(((size_t)b * 16 + h) * 64 + d) * 4096 + l] = (__bf16)val;
        } else {
          Yf[(size_t)row * 1024 + col] = val;
        }
      }
}

// ---------------------------------------------------------------------------
// Landmark means: q_t/k_t [B,H,64,64] = segment means over L/m = 64.
// ---------------------------------------------------------------------------
__global__ __launch_bounds__(256) void nys_landmarks(
    const __bf16* __restrict__ Qh, const __bf16* __restrict__ Kh,
    __bf16* __restrict__ qt, __bf16* __restrict__ kt) {
  int bh = blockIdx.x;
  size_t base = (size_t)bh * 4096 * 64;
  for (int idx = threadIdx.x; idx < 4096; idx += 256) {
    int mi = idx >> 6, d = idx & 63;
    const __bf16* pq = Qh + base + (size_t)(mi * 64) * 64 + d;
    const __bf16* pk = Kh + base + (size_t)(mi * 64) * 64 + d;
    float sq = 0.f, sk = 0.f;
    for (int j = 0; j < 64; ++j) {
      sq += (float)pq[j * 64];
      sk += (float)pk[j * 64];
    }
    qt[(size_t)bh * 4096 + idx] = (__bf16)(sq * (1.0f / 64.0f));
    kt[(size_t)bh * 4096 + idx] = (__bf16)(sk * (1.0f / 64.0f));
  }
}

// ---------------------------------------------------------------------------
// kernel_2 = softmax(q_t @ k_t^T) + per-(b,h) max column-sum / row-sum.
// ---------------------------------------------------------------------------
__global__ __launch_bounds__(256) void nys_kernel2(
    const __bf16* __restrict__ qt, const __bf16* __restrict__ kt,
    float* __restrict__ K2, float* __restrict__ colmax,
    float* __restrict__ rowmax) {
  __shared__ __align__(16) __bf16 qts[64][80];
  __shared__ __align__(16) __bf16 kts[64][80];
  __shared__ float Sld[64][65];
  __shared__ float redc[64], redr[64];

  const int tid = threadIdx.x, lane = tid & 31, wid = tid >> 5;
  const int r = lane & 15, half = lane >> 4;
  const int bh = blockIdx.x;

  for (int i = tid; i < 512; i += 256) {
    int row = i >> 3, c8 = (i & 7) << 3;
    *(v8bf*)&qts[row][c8] = *(const v8bf*)(qt + (size_t)bh * 4096 + row * 64 + c8);
    *(v8bf*)&kts[row][c8] = *(const v8bf*)(kt + (size_t)bh * 4096 + row * 64 + c8);
  }
  __syncthreads();

  for (int u = 0; u < 2; ++u) {
    int t = wid * 2 + u, tm = t >> 2, tn = t & 3;
    v8f c = ZV8;
    for (int k0 = 0; k0 < 64; k0 += 32) {
      v16bf fa = frag_ld(&qts[tm * 16][k0], 80, r, half);
      v16bf fb = frag_ld(&kts[tn * 16][k0], 80, r, half);
      c = wmma_bf16(fa, fb, c);
    }
    for (int vv = 0; vv < 8; ++vv)
      Sld[tm * 16 + half * 8 + vv][tn * 16 + r] = c[vv];
  }
  __syncthreads();

  if (tid < 64) {
    float m = -3.0e38f;
    for (int c = 0; c < 64; ++c) m = fmaxf(m, Sld[tid][c]);
    float s = 0.f;
    for (int c = 0; c < 64; ++c) s += __expf(Sld[tid][c] - m);
    float inv = 1.0f / s, rs = 0.f;
    for (int c = 0; c < 64; ++c) {
      float p = __expf(Sld[tid][c] - m) * inv;
      Sld[tid][c] = p;
      rs += p;
      K2[(size_t)bh * 4096 + tid * 64 + c] = p;
    }
    redr[tid] = rs;
  }
  __syncthreads();
  if (tid < 64) {
    float cs = 0.f;
    for (int i = 0; i < 64; ++i) cs += Sld[i][tid];
    redc[tid] = cs;
  }
  __syncthreads();
  if (tid == 0) {
    float cm = 0.f, rm = 0.f;
    for (int i = 0; i < 64; ++i) {
      cm = fmaxf(cm, redc[i]);
      rm = fmaxf(rm, redr[i]);
    }
    colmax[bh] = cm;
    rowmax[bh] = rm;
  }
}

// ---------------------------------------------------------------------------
// Newton-Schulz pseudo-inverse of kernel_2 (64x64, f32 WMMA).
// ---------------------------------------------------------------------------
__device__ inline void mm64(float (*D)[65], const float (*A)[65],
                            const float (*B)[65], int tid) {
  __syncthreads();
  int lane = tid & 31, wid = tid >> 5, r = lane & 15, half = lane >> 4;
  for (int u = 0; u < 2; ++u) {
    int t = wid * 2 + u, tm = t >> 2, tn = t & 3;
    v8f acc = ZV8;
    for (int k0 = 0; k0 < 64; k0 += 4) {
      v2f av, bv;
      av[0] = A[tm * 16 + r][k0 + half * 2 + 0];
      av[1] = A[tm * 16 + r][k0 + half * 2 + 1];
      bv[0] = B[k0 + half * 2 + 0][tn * 16 + r];
      bv[1] = B[k0 + half * 2 + 1][tn * 16 + r];
      acc = wmma_f32(av, bv, acc);
    }
    for (int vv = 0; vv < 8; ++vv)
      D[tm * 16 + half * 8 + vv][tn * 16 + r] = acc[vv];
  }
  __syncthreads();
}

__global__ __launch_bounds__(256) void nys_pinv(
    const float* __restrict__ K2, const float* __restrict__ colmax,
    const float* __restrict__ rowmax, __bf16* __restrict__ pinvb) {
  __shared__ float Am[64][65], Zm[64][65], AZ[64][65], P1[64][65], P2[64][65];
  const int tid = threadIdx.x, bh = blockIdx.x;

  for (int i = tid; i < 4096; i += 256)
    Am[i >> 6][i & 63] = K2[(size_t)bh * 4096 + i];
  float cm = 0.f, rm = 0.f;
  for (int i = 0; i < 64; ++i) {
    cm = fmaxf(cm, colmax[i]);
    rm = fmaxf(rm, rowmax[i]);
  }
  const float inv = 1.0f / (cm * rm);
  __syncthreads();
  for (int i = tid; i < 4096; i += 256) {
    int ii = i >> 6, jj = i & 63;
    Zm[ii][jj] = Am[jj][ii] * inv;  // z = a^T / denom
  }

  for (int it = 0; it < 6; ++it) {
    mm64(AZ, Am, Zm, tid);
    for (int i = tid; i < 4096; i += 256) {
      int ii = i >> 6, jj = i & 63;
      P1[ii][jj] = (ii == jj ? 7.0f : 0.0f) - AZ[ii][jj];
    }
    mm64(P2, AZ, P1, tid);
    for (int i = tid; i < 4096; i += 256) {
      int ii = i >> 6, jj = i & 63;
      P1[ii][jj] = (ii == jj ? 15.0f : 0.0f) - P2[ii][jj];
    }
    mm64(P2, AZ, P1, tid);
    for (int i = tid; i < 4096; i += 256) {
      int ii = i >> 6, jj = i & 63;
      P1[ii][jj] = (ii == jj ? 13.0f : 0.0f) - P2[ii][jj];
    }
    mm64(P2, Zm, P1, tid);
    for (int i = tid; i < 4096; i += 256) {
      int ii = i >> 6, jj = i & 63;
      Zm[ii][jj] = 0.25f * P2[ii][jj];
    }
  }
  __syncthreads();
  for (int i = tid; i < 4096; i += 256)
    pinvb[(size_t)bh * 4096 + i] = (__bf16)Zm[i >> 6][i & 63];
}

// ---------------------------------------------------------------------------
// kernel_1 = softmax(q @ k_t^T) -> bf16 [B,H,L,64].  Q tile staged via TDM.
// ---------------------------------------------------------------------------
__global__ __launch_bounds__(256) void nys_kernel1(
    const __bf16* __restrict__ Qh, const __bf16* __restrict__ kt,
    __bf16* __restrict__ K1) {
  __shared__ __align__(16) __bf16 Qs[128][80];
  __shared__ __align__(16) __bf16 kts[64][80];
  __shared__ float Sld[128][65];

  const int tid = threadIdx.x, lane = tid & 31, wid = tid >> 5;
  const int r = lane & 15, half = lane >> 4;
  const int bh = blockIdx.y, l0 = blockIdx.x * 128;

  stage_rows<64>(&Qs[0][0], Qh + ((size_t)bh * 4096 + l0) * 64, 64, 128, tid);
  for (int i = tid; i < 512; i += 256) {
    int row = i >> 3, c8 = (i & 7) << 3;
    *(v8bf*)&kts[row][c8] = *(const v8bf*)(kt + (size_t)bh * 4096 + row * 64 + c8);
  }
  stage_wait(tid);
  __syncthreads();

  v8f acc[4];
  for (int j = 0; j < 4; ++j) acc[j] = ZV8;
  for (int k0 = 0; k0 < 64; k0 += 32) {
    v16bf fa = frag_ld(&Qs[wid * 16][k0], 80, r, half);
    for (int j = 0; j < 4; ++j) {
      v16bf fb = frag_ld(&kts[j * 16][k0], 80, r, half);
      acc[j] = wmma_bf16(fa, fb, acc[j]);
    }
  }
  for (int j = 0; j < 4; ++j)
    for (int vv = 0; vv < 8; ++vv)
      Sld[wid * 16 + half * 8 + vv][j * 16 + r] = acc[j][vv];
  __syncthreads();

  if (tid < 128) {
    float m = -3.0e38f;
    for (int c = 0; c < 64; ++c) m = fmaxf(m, Sld[tid][c]);
    float s = 0.f;
    for (int c = 0; c < 64; ++c) s += __expf(Sld[tid][c] - m);
    float inv = 1.0f / s;
    __bf16* out = K1 + ((size_t)bh * 4096 + l0 + tid) * 64;
    for (int c = 0; c < 64; ++c)
      out[c] = (__bf16)(__expf(Sld[tid][c] - m) * inv);
  }
}

// ---------------------------------------------------------------------------
// Flash-style t1 = softmax(q_t @ k^T - mask) @ v, then t2 = pinv @ t1 stored
// transposed [B,H,D,64].  One block per (b,h); K chunks staged via TDM,
// software-pipelined across the softmax / P@V stages.
// ---------------------------------------------------------------------------
__global__ __launch_bounds__(256) void nys_flash(
    const __bf16* __restrict__ qt, const __bf16* __restrict__ Kh,
    const __bf16* __restrict__ Vt, const unsigned char* __restrict__ mask,
    const __bf16* __restrict__ pinvb, __bf16* __restrict__ t2t) {
  __shared__ __align__(16) __bf16 qts[64][80];
  __shared__ __align__(16) __bf16 Ks[128][80];
  __shared__ float Sld[64][132];
  __shared__ __align__(16) __bf16 P[64][136];
  __shared__ float mstate[64], lstate[64], sstate[64];
  __shared__ __align__(16) __bf16 t1t[64][80];

  const int tid = threadIdx.x, lane = tid & 31, wid = tid >> 5;
  const int r = lane & 15, half = lane >> 4;
  const int bh = blockIdx.x, b = bh >> 4;

  stage_rows<64>(&Ks[0][0], Kh + (size_t)bh * 4096 * 64, 64, 128, tid);
  for (int i = tid; i < 512; i += 256) {
    int row = i >> 3, c8 = (i & 7) << 3;
    *(v8bf*)&qts[row][c8] = *(const v8bf*)(qt + (size_t)bh * 4096 + row * 64 + c8);
  }
  if (tid < 64) {
    mstate[tid] = -3.0e38f;
    lstate[tid] = 0.f;
  }
  v8f acc2[2];
  acc2[0] = ZV8;
  acc2[1] = ZV8;

  for (int lc = 0; lc < 4096; lc += 128) {
    stage_wait(tid);
    __syncthreads();

    // S = q_t @ k_chunk^T  (64 x 128), masked
    for (int u = 0; u < 4; ++u) {
      int t = wid * 4 + u, tm = t >> 3, tn = t & 7;
      v8f c = ZV8;
      for (int k0 = 0; k0 < 64; k0 += 32) {
        v16bf fa = frag_ld(&qts[tm * 16][k0], 80, r, half);
        v16bf fb = frag_ld(&Ks[tn * 16][k0], 80, r, half);
        c = wmma_bf16(fa, fb, c);
      }
      int coln = tn * 16 + r;
      float pen = mask[b * 4096 + lc + coln] ? 0.0f : -1.0e9f;
      for (int vv = 0; vv < 8; ++vv)
        Sld[tm * 16 + half * 8 + vv][coln] = c[vv] + pen;
    }
    __syncthreads();

    // Ks fully consumed: overlap next chunk's TDM with softmax + P@V below
    if (lc + 128 < 4096)
      stage_rows<64>(&Ks[0][0], Kh + ((size_t)bh * 4096 + lc + 128) * 64, 64,
                     128, tid);

    if (tid < 64) {  // online softmax stats per row
      float mold = mstate[tid], rm = mold;
      for (int c = 0; c < 128; ++c) rm = fmaxf(rm, Sld[tid][c]);
      float sc = __expf(mold - rm), s = 0.f;
      for (int c = 0; c < 128; ++c) {
        float e = __expf(Sld[tid][c] - rm);
        s += e;
        P[tid][c] = (__bf16)e;
      }
      lstate[tid] = lstate[tid] * sc + s;
      mstate[tid] = rm;
      sstate[tid] = sc;
    }
    __syncthreads();

    // O = O*scale + P @ v_chunk  (v pre-transposed: Vt[d][l], l contiguous)
    for (int u = 0; u < 2; ++u) {
      int t = wid * 2 + u, tm = t >> 2, tn = t & 3;
      for (int vv = 0; vv < 8; ++vv) acc2[u][vv] *= sstate[tm * 16 + half * 8 + vv];
      for (int k0 = 0; k0 < 128; k0 += 32) {
        v16bf fa = frag_ld(&P[tm * 16][k0], 136, r, half);
        v16bf fb = frag_ld(Vt + ((size_t)bh * 64 + tn * 16) * 4096 + lc + k0,
                           4096, r, half);
        acc2[u] = wmma_bf16(fa, fb, acc2[u]);
      }
    }
    __syncthreads();
  }
  if (tid < 64) lstate[tid] = 1.0f / lstate[tid];
  __syncthreads();

  // t1 normalized, stored transposed (t1t[d][m]) -> K-contiguous B operand
  for (int u = 0; u < 2; ++u) {
    int t = wid * 2 + u, tm = t >> 2, tn = t & 3;
    int col = tn * 16 + r;
    for (int vv = 0; vv < 8; ++vv) {
      int row = tm * 16 + half * 8 + vv;
      t1t[col][row] = (__bf16)(acc2[u][vv] * lstate[row]);
    }
  }
  __syncthreads();

  // t2 = pinv @ t1, written transposed t2t[B,H,D,64]
  for (int u = 0; u < 2; ++u) {
    int t = wid * 2 + u, tm = t >> 2, tn = t & 3;
    v8f c = ZV8;
    for (int k0 = 0; k0 < 64; k0 += 32) {
      v16bf fa = frag_ld(pinvb + (size_t)bh * 4096 + (tm * 16) * 64 + k0, 64, r, half);
      v16bf fb = frag_ld(&t1t[tn * 16][k0], 80, r, half);
      c = wmma_bf16(fa, fb, c);
    }
    for (int vv = 0; vv < 8; ++vv) {
      int row = tm * 16 + half * 8 + vv;  // m index
      int col = tn * 16 + r;              // d index
      t2t[(size_t)bh * 4096 + col * 64 + row] = (__bf16)c[vv];
    }
  }
}

// ---------------------------------------------------------------------------
// out_heads = kernel_1 @ t2 -> bf16 [B,L,1024].  K1 tile staged via TDM.
// ---------------------------------------------------------------------------
__global__ __launch_bounds__(256) void nys_outhead(
    const __bf16* __restrict__ K1, const __bf16* __restrict__ t2t,
    __bf16* __restrict__ OH) {
  __shared__ __align__(16) __bf16 K1s[128][80];
  __shared__ __align__(16) __bf16 T2s[64][80];

  const int tid = threadIdx.x, lane = tid & 31, wid = tid >> 5;
  const int r = lane & 15, half = lane >> 4;
  const int bh = blockIdx.y, l0 = blockIdx.x * 128;
  const int b = bh >> 4, h = bh & 15;

  stage_rows<64>(&K1s[0][0], K1 + ((size_t)bh * 4096 + l0) * 64, 64, 128, tid);
  for (int i = tid; i < 512; i += 256) {
    int row = i >> 3, c8 = (i & 7) << 3;
    *(v8bf*)&T2s[row][c8] = *(const v8bf*)(t2t + (size_t)bh * 4096 + row * 64 + c8);
  }
  stage_wait(tid);
  __syncthreads();

  v8f acc[4];
  for (int j = 0; j < 4; ++j) acc[j] = ZV8;
  for (int k0 = 0; k0 < 64; k0 += 32) {
    v16bf fa = frag_ld(&K1s[wid * 16][k0], 80, r, half);
    for (int j = 0; j < 4; ++j) {
      v16bf fb = frag_ld(&T2s[j * 16][k0], 80, r, half);
      acc[j] = wmma_bf16(fa, fb, acc[j]);
    }
  }
  for (int j = 0; j < 4; ++j)
    for (int vv = 0; vv < 8; ++vv) {
      int l = l0 + wid * 16 + half * 8 + vv;
      int d = j * 16 + r;
      OH[((size_t)b * 4096 + l) * 1024 + h * 64 + d] = (__bf16)acc[j][vv];
    }
}

// ---------------------------------------------------------------------------
// Host side
// ---------------------------------------------------------------------------
static inline size_t align256(size_t x) { return (x + 255) & ~(size_t)255; }

extern "C" void kernel_launch(void* const* d_in, const int* in_sizes, int n_in,
                              void* d_out, int out_size, void* d_ws,
                              size_t ws_size, hipStream_t stream) {
  const float* q = (const float*)d_in[0];
  const float* k = (const float*)d_in[1];
  const float* v = (const float*)d_in[2];
  const unsigned char* mask = (const unsigned char*)d_in[3];
  const float* Wq = (const float*)d_in[4];
  const float* bq = (const float*)d_in[5];
  const float* Wk = (const float*)d_in[6];
  const float* bk = (const float*)d_in[7];
  const float* Wv = (const float*)d_in[8];
  const float* bv = (const float*)d_in[9];
  const float* Wo = (const float*)d_in[10];
  const float* bo = (const float*)d_in[11];

  char* w = (char*)d_ws;
  size_t off = 0;
  const size_t big = (size_t)4 * 16 * 4096 * 64;  // 16.7M elems
  __bf16* qb  = (__bf16*)(w + off); off = align256(off + big * 2);
  __bf16* kb  = (__bf16*)(w + off); off = align256(off + big * 2);
  __bf16* vb  = (__bf16*)(w + off); off = align256(off + big * 2);
  __bf16* Wqb = (__bf16*)(w + off); off = align256(off + 1048576 * 2);
  __bf16* Wkb = (__bf16*)(w + off); off = align256(off + 1048576 * 2);
  __bf16* Wvb = (__bf16*)(w + off); off = align256(off + 1048576 * 2);
  __bf16* Wob = (__bf16*)(w + off); off = align256(off + 1048576 * 2);
  __bf16* Qh  = (__bf16*)(w + off); off = align256(off + big * 2);
  __bf16* Kh  = (__bf16*)(w + off); off = align256(off + big * 2);
  __bf16* Vt  = (__bf16*)(w + off); off = align256(off + big * 2);
  __bf16* qt  = (__bf16*)(w + off); off = align256(off + 262144 * 2);
  __bf16* kt  = (__bf16*)(w + off); off = align256(off + 262144 * 2);
  float*  K2  = (float*)(w + off);  off = align256(off + 262144 * 4);
  float*  cmx = (float*)(w + off);  off = align256(off + 64 * 4);
  float*  rmx = (float*)(w + off);  off = align256(off + 64 * 4);
  __bf16* pinvb = (__bf16*)(w + off); off = align256(off + 262144 * 2);
  __bf16* K1  = (__bf16*)(w + off); off = align256(off + big * 2);
  __bf16* t2t = (__bf16*)(w + off); off = align256(off + 262144 * 2);
  __bf16* OH  = (__bf16*)(w + off); off = align256(off + big * 2);

  dim3 blk(256);
  to_bf16<<<16384, blk, 0, stream>>>(q, qb, 4194304);
  to_bf16<<<16384, blk, 0, stream>>>(k, kb, 4194304);
  to_bf16<<<16384, blk, 0, stream>>>(v, vb, 4194304);
  to_bf16<<<1024, blk, 0, stream>>>(Wq, Wqb, 262144);
  to_bf16<<<1024, blk, 0, stream>>>(Wk, Wkb, 262144);
  to_bf16<<<1024, blk, 0, stream>>>(Wv, Wvb, 262144);
  to_bf16<<<1024, blk, 0, stream>>>(Wo, Wob, 262144);

  nys_gemm<0><<<dim3(128, 8), blk, 0, stream>>>(qb, Wqb, bq, mask, Qh, nullptr);
  nys_gemm<1><<<dim3(128, 8), blk, 0, stream>>>(kb, Wkb, bk, mask, Kh, nullptr);
  nys_gemm<2><<<dim3(128, 8), blk, 0, stream>>>(vb, Wvb, bv, nullptr, Vt, nullptr);
  nys_landmarks<<<64, blk, 0, stream>>>(Qh, Kh, qt, kt);
  nys_kernel2<<<64, blk, 0, stream>>>(qt, kt, K2, cmx, rmx);
  nys_pinv<<<64, blk, 0, stream>>>(K2, cmx, rmx, pinvb);
  nys_kernel1<<<dim3(32, 64), blk, 0, stream>>>(Qh, kt, K1);
  nys_flash<<<64, blk, 0, stream>>>(qt, Kh, Vt, mask, pinvb, t2t);
  nys_outhead<<<dim3(32, 64), blk, 0, stream>>>(K1, t2t, OH);
  nys_gemm<3><<<dim3(128, 8), blk, 0, stream>>>(OH, Wob, bo, nullptr, nullptr,
                                                (float*)d_out);
}
